// PointTransformerLayer_75093208203616
// MI455X (gfx1250) — matrix operand order repaired and checked by reading, not compiled, not run on hardware
//
#include <hip/hip_runtime.h>

// ---------------------------------------------------------------------------
// Point Transformer layer for MI455X (gfx1250, wave32, WMMA bf16 16x16x32).
// N=65536 points, NS=16 neighbors, C=256 channels, MID/SHARE=32.
// All GEMMs in bf16 WMMA with f32 accumulation; gathered tensors stored bf16
// to halve HBM traffic (gathers dominate the roofline at 23.3 TB/s).
// BatchNorm stats via deterministic two-level reductions (no atomics).
// ---------------------------------------------------------------------------

typedef __attribute__((ext_vector_type(16))) __bf16 v16bf;
typedef __attribute__((ext_vector_type(8)))  __bf16 v8bf;
typedef __attribute__((ext_vector_type(8)))  float  v8f;
typedef __attribute__((ext_vector_type(4)))  float  v4f;

#define NPTS   65536
#define NS     16
#define CCH    256
#define QKVW   768
#define MIDS   32
#define CNTF   1048576.0f   // N*NS
#define EPSBN  1e-5f

// workspace layout (bytes, all 256B aligned)
#define OFF_QKV   ((size_t)0)                       // bf16 [N][768]
#define OFF_H1    ((size_t)100663296)               // f32  [N*16][32]
#define OFF_WB    ((size_t)(OFF_H1   + 134217728))  // bf16 [768][256]
#define OFF_BQKV  ((size_t)(OFF_WB   + 393216))     // f32  [768]
#define OFF_WW1   ((size_t)(OFF_BQKV + 3072))       // bf16 [32][256]
#define OFF_WW2   ((size_t)(OFF_WW1  + 16384))      // bf16 [32][32]
#define OFF_STATS ((size_t)(OFF_WW2  + 2048))       // f32  582 (pad 4096)
#define OFF_PARTP ((size_t)(OFF_STATS + 4096))      // f32  2048*6
#define OFF_PART1 ((size_t)(OFF_PARTP + 49152))     // f32  2048*512
#define OFF_PART2 ((size_t)(OFF_PART1 + 4194304))   // f32  1024*64
// stats layout: [0..2] bnp_sum, [3..5] bnp_sq, [6..261] bn1_sum,
//               [262..517] bn1_sq, [518..549] bn2_sum, [550..581] bn2_sq

__device__ __forceinline__ v8f wmma_bf16(v16bf a, v16bf b, v8f c) {
  return __builtin_amdgcn_wmma_f32_16x16x32_bf16(
      /*neg_a=*/false, a, /*neg_b=*/false, b,
      /*c_mod=*/(short)0, c, /*reuse_a=*/false, /*reuse_b=*/false);
}

// post-BNp ReLU 3-vector for one (point, neighbor) pair
__device__ __forceinline__ void compute_rb(
    const float* __restrict__ p, int n, int j,
    const float* __restrict__ wp1, const float* __restrict__ bp1,
    const float* sp, const float* sh, float* out3) {
  float d0 = p[j * 3 + 0] - p[n * 3 + 0];
  float d1 = p[j * 3 + 1] - p[n * 3 + 1];
  float d2 = p[j * 3 + 2] - p[n * 3 + 2];
#pragma unroll
  for (int r = 0; r < 3; ++r) {
    float u = wp1[r * 3 + 0] * d0 + wp1[r * 3 + 1] * d1 + wp1[r * 3 + 2] * d2 + bp1[r];
    u = sp[r] * u + sh[r];
    out3[r] = fmaxf(u, 0.0f);
  }
}

// ---------------------------------------------------------------------------
// K0: pack weights to bf16 (fused QKV [768][256], ww1, ww2) + fused bias
// ---------------------------------------------------------------------------
__global__ __launch_bounds__(256) void k0_prep(
    const float* __restrict__ wq, const float* __restrict__ wk, const float* __restrict__ wv,
    const float* __restrict__ bq, const float* __restrict__ bk, const float* __restrict__ bv,
    const float* __restrict__ ww1, const float* __restrict__ ww2,
    __bf16* __restrict__ wb, float* __restrict__ bqkv,
    __bf16* __restrict__ wwb1, __bf16* __restrict__ wwb2) {
  int i = blockIdx.x * 256 + threadIdx.x;
  if (i < 768 * 256) {
    int row = i >> 8, k = i & 255;
    float v = (row < 256) ? wq[row * 256 + k]
            : (row < 512) ? wk[(row - 256) * 256 + k]
                          : wv[(row - 512) * 256 + k];
    wb[i] = (__bf16)v;
  }
  if (i < 768) bqkv[i] = (i < 256) ? bq[i] : (i < 512) ? bk[i - 256] : bv[i - 512];
  if (i < 32 * 256) wwb1[i] = (__bf16)ww1[i];
  if (i < 32 * 32)  wwb2[i] = (__bf16)ww2[i];
}

// ---------------------------------------------------------------------------
// K1: QKV GEMM  qkv[n][m] = sum_k x[n][k] * Wfused[m][k] + b[m]  (bf16 WMMA)
// One wave computes a 16-row x 128-col tile. grid = (512, 6), block = 256.
// ---------------------------------------------------------------------------
__global__ __launch_bounds__(256) void k1_qkv(
    const float* __restrict__ x, const __bf16* __restrict__ wb,
    const float* __restrict__ bqkv, __bf16* __restrict__ qkv) {
  const int lane = threadIdx.x & 31;
  const int wave = threadIdx.x >> 5;
  const int rowbase = (blockIdx.x * 8 + wave) * 16;
  const int colbase = blockIdx.y * 128;
  const int m  = lane & 15;
  const int kb = (lane < 16) ? 0 : 8;
  const int hi = (lane < 16) ? 0 : 8;
  const int ko = (lane < 16) ? 0 : 16;
  const float* xrow = x + (size_t)(rowbase + m) * CCH;
  const __bf16* bptr[8];
#pragma unroll
  for (int t = 0; t < 8; ++t)
    bptr[t] = wb + (size_t)(colbase + t * 16 + m) * CCH + ko;
  v8f acc[8] = {};
#pragma unroll
  for (int k0 = 0; k0 < CCH; k0 += 32) {
    v4f f0 = *(const v4f*)(xrow + k0 + ko);
    v4f f1 = *(const v4f*)(xrow + k0 + ko + 4);
    v4f f2 = *(const v4f*)(xrow + k0 + ko + 8);
    v4f f3 = *(const v4f*)(xrow + k0 + ko + 12);
    v16bf a;
#pragma unroll
    for (int i = 0; i < 4; ++i) {
      a[i]      = (__bf16)f0[i];
      a[i + 4]  = (__bf16)f1[i];
      a[i + 8]  = (__bf16)f2[i];
      a[i + 12] = (__bf16)f3[i];
    }
#pragma unroll
    for (int t = 0; t < 8; ++t) {
      v16bf b = *(const v16bf*)(bptr[t] + k0);
      acc[t] = wmma_bf16(a, b, acc[t]);
    }
  }
#pragma unroll
  for (int t = 0; t < 8; ++t) {
    int col = colbase + t * 16 + m;
    float bias = bqkv[col];
#pragma unroll
    for (int v = 0; v < 8; ++v) {
      int row = rowbase + v + hi;
      qkv[(size_t)row * QKVW + col] = (__bf16)(acc[t][v] + bias);
    }
  }
}

// ---------------------------------------------------------------------------
// K2: BN(p) partial stats of u = (p[idx]-p) @ wp1^T + bp1  (3 channels)
// ---------------------------------------------------------------------------
__global__ __launch_bounds__(256) void k2_bnp_part(
    const float* __restrict__ p, const int* __restrict__ idx,
    const float* __restrict__ wp1, const float* __restrict__ bp1,
    float* __restrict__ part) {
  float s[3] = {0.f, 0.f, 0.f}, q[3] = {0.f, 0.f, 0.f};
  int gid = blockIdx.x * 256 + threadIdx.x;
  int stride = gridDim.x * 256;
  for (int i = gid; i < NPTS * NS; i += stride) {
    int n = i >> 4;
    int j = idx[i];
    float d0 = p[j * 3 + 0] - p[n * 3 + 0];
    float d1 = p[j * 3 + 1] - p[n * 3 + 1];
    float d2 = p[j * 3 + 2] - p[n * 3 + 2];
#pragma unroll
    for (int r = 0; r < 3; ++r) {
      float u = wp1[r * 3 + 0] * d0 + wp1[r * 3 + 1] * d1 + wp1[r * 3 + 2] * d2 + bp1[r];
      s[r] += u;
      q[r] += u * u;
    }
  }
  __shared__ float red[256][6];
#pragma unroll
  for (int r = 0; r < 3; ++r) {
    red[threadIdx.x][r] = s[r];
    red[threadIdx.x][3 + r] = q[r];
  }
  __syncthreads();
  if (threadIdx.x < 6) {
    float t = 0.f;
    for (int r = 0; r < 256; ++r) t += red[r][threadIdx.x];
    part[(size_t)blockIdx.x * 6 + threadIdx.x] = t;
  }
}

// generic deterministic column reduce: dst[t] = sum_b part[b*W + t]
__global__ void kreduce(const float* __restrict__ part, float* __restrict__ dst,
                        int W, int B) {
  int t = blockIdx.x * blockDim.x + threadIdx.x;
  if (t < W) {
    float s = 0.f;
    for (int b = 0; b < B; ++b) s += part[(size_t)b * W + t];
    dst[t] = s;
  }
}

// ---------------------------------------------------------------------------
// K3: BN1 partial stats of w_pre = xk[idx] - xq + p_r  (256 channels)
// 32 points per block, one thread per channel. Deterministic partials.
// ---------------------------------------------------------------------------
__global__ __launch_bounds__(256) void k3_bn1_part(
    const float* __restrict__ p, const int* __restrict__ idx,
    const __bf16* __restrict__ qkv,
    const float* __restrict__ wp1, const float* __restrict__ bp1,
    const float* __restrict__ gp, const float* __restrict__ btp,
    const float* __restrict__ wp2, const float* __restrict__ bp2,
    const float* __restrict__ stats, float* __restrict__ part) {
  __shared__ float sp[3], sh[3];
  __shared__ float rb[32][NS][3];
  __shared__ int   sidx[32][NS];
  const int t = threadIdx.x;
  const int n0 = blockIdx.x * 32;
  if (t < 3) {
    float mean = stats[t] * (1.0f / CNTF);
    float var  = stats[3 + t] * (1.0f / CNTF) - mean * mean;
    float inv  = rsqrtf(var + EPSBN);
    sp[t] = gp[t] * inv;
    sh[t] = btp[t] - mean * sp[t];
  }
  for (int i = t; i < 32 * NS; i += 256)
    sidx[i >> 4][i & 15] = idx[(size_t)(n0 + (i >> 4)) * NS + (i & 15)];
  __syncthreads();
  for (int i = t; i < 32 * NS; i += 256) {
    int pt = i >> 4, s = i & 15;
    compute_rb(p, n0 + pt, sidx[pt][s], wp1, bp1, sp, sh, &rb[pt][s][0]);
  }
  __syncthreads();
  const int c = t;
  const float w20 = wp2[c * 3 + 0], w21 = wp2[c * 3 + 1], w22 = wp2[c * 3 + 2];
  const float b2c = bp2[c];
  float sum = 0.f, sq = 0.f;
  for (int pt = 0; pt < 32; ++pt) {
    int n = n0 + pt;
    float xq = (float)qkv[(size_t)n * QKVW + c];
#pragma unroll
    for (int s = 0; s < NS; ++s) {
      int j = sidx[pt][s];
      float xk = (float)qkv[(size_t)j * QKVW + CCH + c];
      float pr = rb[pt][s][0] * w20 + rb[pt][s][1] * w21 + rb[pt][s][2] * w22 + b2c;
      float v = xk - xq + pr;
      sum += v;
      sq += v * v;
    }
  }
  part[(size_t)blockIdx.x * 512 + c] = sum;
  part[(size_t)blockIdx.x * 512 + 256 + c] = sq;
}

// ---------------------------------------------------------------------------
// K4: h1 = relu(bn1(w_pre)) @ ww1^T + bw1   -> f32 [N*16][32]
// Block = 8 points. Phase 1: fused gather+BN1+ReLU into LDS bf16 tile.
// Phase 2: one wave per point, WMMA from LDS (ds_load_b128 A-frags).
// ---------------------------------------------------------------------------
__global__ __launch_bounds__(256) void k4_h1(
    const float* __restrict__ p, const int* __restrict__ idx,
    const __bf16* __restrict__ qkv,
    const float* __restrict__ wp1, const float* __restrict__ bp1,
    const float* __restrict__ gp, const float* __restrict__ btp,
    const float* __restrict__ wp2, const float* __restrict__ bp2,
    const float* __restrict__ g1, const float* __restrict__ b1,
    const __bf16* __restrict__ wwb1, const float* __restrict__ bw1,
    const float* __restrict__ stats, float* __restrict__ h1) {
  __shared__ float sp[3], sh[3];
  __shared__ float rb[8][NS][3];
  __shared__ int   sidx[8][NS];
  __shared__ __bf16 wt[8][NS][CCH];   // 64 KB
  const int t = threadIdx.x;
  const int n0 = blockIdx.x * 8;
  if (t < 3) {
    float mean = stats[t] * (1.0f / CNTF);
    float var  = stats[3 + t] * (1.0f / CNTF) - mean * mean;
    float inv  = rsqrtf(var + EPSBN);
    sp[t] = gp[t] * inv;
    sh[t] = btp[t] - mean * sp[t];
  }
  if (t < 128) sidx[t >> 4][t & 15] = idx[(size_t)(n0 + (t >> 4)) * NS + (t & 15)];
  __syncthreads();
  if (t < 128) compute_rb(p, n0 + (t >> 4), sidx[t >> 4][t & 15], wp1, bp1, sp, sh,
                          &rb[t >> 4][t & 15][0]);
  __syncthreads();
  // phase 1: one thread per channel
  const int c = t;
  float mean1 = stats[6 + c] * (1.0f / CNTF);
  float var1  = stats[262 + c] * (1.0f / CNTF) - mean1 * mean1;
  float inv1  = rsqrtf(var1 + EPSBN);
  float sc1 = g1[c] * inv1;
  float sf1 = b1[c] - mean1 * sc1;
  const float w20 = wp2[c * 3 + 0], w21 = wp2[c * 3 + 1], w22 = wp2[c * 3 + 2];
  const float b2c = bp2[c];
  for (int pt = 0; pt < 8; ++pt) {
    int n = n0 + pt;
    float xq = (float)qkv[(size_t)n * QKVW + c];
#pragma unroll
    for (int s = 0; s < NS; ++s) {
      int j = sidx[pt][s];
      float xk = (float)qkv[(size_t)j * QKVW + CCH + c];
      float pr = rb[pt][s][0] * w20 + rb[pt][s][1] * w21 + rb[pt][s][2] * w22 + b2c;
      float v = xk - xq + pr;
      wt[pt][s][c] = (__bf16)fmaxf(sc1 * v + sf1, 0.0f);
    }
  }
  __syncthreads();
  // phase 2: WMMA (16 s) x (256 k) @ (256 k) x (32 out)
  const int lane = t & 31;
  const int wave = t >> 5;
  const int m  = lane & 15;
  const int kb = (lane < 16) ? 0 : 8;
  const int hi = (lane < 16) ? 0 : 8;
  const int ko = (lane < 16) ? 0 : 16;
  v8f c0 = {}, c1 = {};
  const __bf16* arow = &wt[wave][m][0];
#pragma unroll
  for (int k0 = 0; k0 < CCH; k0 += 32) {
    v8bf lo  = *(const v8bf*)(arow + k0 + ko);
    v8bf hi8 = *(const v8bf*)(arow + k0 + ko + 8);
    v16bf a = __builtin_shufflevector(lo, hi8, 0, 1, 2, 3, 4, 5, 6, 7,
                                      8, 9, 10, 11, 12, 13, 14, 15);
    v16bf b0 = *(const v16bf*)(wwb1 + (size_t)m * CCH + k0 + ko);
    v16bf b1v = *(const v16bf*)(wwb1 + (size_t)(m + 16) * CCH + k0 + ko);
    c0 = wmma_bf16(a, b0, c0);
    c1 = wmma_bf16(a, b1v, c1);
  }
  const int n = n0 + wave;
  const float bj0 = bw1[m], bj1 = bw1[m + 16];
#pragma unroll
  for (int v = 0; v < 8; ++v) {
    int s = v + hi;
    size_t base = ((size_t)n * NS + s) * MIDS;
    h1[base + m] = c0[v] + bj0;
    h1[base + 16 + m] = c1[v] + bj1;
  }
}

// ---------------------------------------------------------------------------
// K4b: BN2 partial stats over h1 [N*16][32]
// ---------------------------------------------------------------------------
__global__ __launch_bounds__(256) void k4b_bn2_part(
    const float* __restrict__ h1, float* __restrict__ part) {
  __shared__ float red[8][64];
  const int t = threadIdx.x;
  const int c = t & 31, g = t >> 5;
  float sum = 0.f, sq = 0.f;
  int stride = gridDim.x * 8;
  for (int r = blockIdx.x * 8 + g; r < NPTS * NS; r += stride) {
    float v = h1[(size_t)r * MIDS + c];
    sum += v;
    sq += v * v;
  }
  red[g][c] = sum;
  red[g][32 + c] = sq;
  __syncthreads();
  if (t < 64) {
    float s = 0.f;
#pragma unroll
    for (int gg = 0; gg < 8; ++gg) s += red[gg][t];
    part[(size_t)blockIdx.x * 64 + t] = s;
  }
}

// ---------------------------------------------------------------------------
// K5: h2 = relu(bn2(h1)) @ ww2^T + bw2 ; softmax over s ; weighted gather sum
// Block = 8 points, one wave per point.
// ---------------------------------------------------------------------------
__global__ __launch_bounds__(256) void k5_out(
    const float* __restrict__ p, const int* __restrict__ idx,
    const __bf16* __restrict__ qkv,
    const float* __restrict__ wp1, const float* __restrict__ bp1,
    const float* __restrict__ gp, const float* __restrict__ btp,
    const float* __restrict__ wp2, const float* __restrict__ bp2,
    const float* __restrict__ g2, const float* __restrict__ b2,
    const __bf16* __restrict__ wwb2, const float* __restrict__ bw2,
    const float* __restrict__ stats, const float* __restrict__ h1,
    float* __restrict__ out) {
  __shared__ float sp[3], sh[3];
  __shared__ float sc2[32], sf2[32];
  __shared__ float rb[8][NS][3];
  __shared__ int   sidx[8][NS];
  __shared__ float wsm[8][NS][32];
  const int t = threadIdx.x;
  const int n0 = blockIdx.x * 8;
  if (t < 3) {
    float mean = stats[t] * (1.0f / CNTF);
    float var  = stats[3 + t] * (1.0f / CNTF) - mean * mean;
    float inv  = rsqrtf(var + EPSBN);
    sp[t] = gp[t] * inv;
    sh[t] = btp[t] - mean * sp[t];
  }
  if (t < 32) {
    float mean = stats[518 + t] * (1.0f / CNTF);
    float var  = stats[550 + t] * (1.0f / CNTF) - mean * mean;
    float inv  = rsqrtf(var + EPSBN);
    sc2[t] = g2[t] * inv;
    sf2[t] = b2[t] - mean * sc2[t];
  }
  if (t < 128) sidx[t >> 4][t & 15] = idx[(size_t)(n0 + (t >> 4)) * NS + (t & 15)];
  __syncthreads();
  if (t < 128) compute_rb(p, n0 + (t >> 4), sidx[t >> 4][t & 15], wp1, bp1, sp, sh,
                          &rb[t >> 4][t & 15][0]);
  __syncthreads();
  const int lane = t & 31;
  const int wave = t >> 5;
  const int m  = lane & 15;
  const int kb = (lane < 16) ? 0 : 8;
  const int hi = (lane < 16) ? 0 : 8;
  const int ko = (lane < 16) ? 0 : 16;
  const int n = n0 + wave;
  // A fragment: relu(bn2(h1 row)) -> bf16
  const float* hrow = h1 + ((size_t)n * NS + m) * MIDS;
  v16bf a;
#pragma unroll
  for (int i = 0; i < 8; ++i) {
    int k0 = ko + i, k1 = ko + 8 + i;
    a[i]     = (__bf16)fmaxf(sc2[k0] * hrow[k0] + sf2[k0], 0.0f);
    a[8 + i] = (__bf16)fmaxf(sc2[k1] * hrow[k1] + sf2[k1], 0.0f);
  }
  v16bf b0 = *(const v16bf*)(wwb2 + (size_t)m * 32 + ko);
  v16bf b1v = *(const v16bf*)(wwb2 + (size_t)(m + 16) * 32 + ko);
  v8f z0 = {}, z1 = {};
  v8f c0 = wmma_bf16(a, b0, z0);
  v8f c1 = wmma_bf16(a, b1v, z1);
  const float bj0 = bw2[m], bj1 = bw2[m + 16];
  // softmax over s (16 rows per column, split across lane and lane^16)
  float mx0 = -1e30f, mx1 = -1e30f;
#pragma unroll
  for (int v = 0; v < 8; ++v) {
    c0[v] += bj0;
    c1[v] += bj1;
    mx0 = fmaxf(mx0, c0[v]);
    mx1 = fmaxf(mx1, c1[v]);
  }
  mx0 = fmaxf(mx0, __shfl_xor(mx0, 16, 32));
  mx1 = fmaxf(mx1, __shfl_xor(mx1, 16, 32));
  float s0 = 0.f, s1 = 0.f, e0[8], e1[8];
#pragma unroll
  for (int v = 0; v < 8; ++v) {
    e0[v] = __expf(c0[v] - mx0);
    e1[v] = __expf(c1[v] - mx1);
    s0 += e0[v];
    s1 += e1[v];
  }
  s0 += __shfl_xor(s0, 16, 32);
  s1 += __shfl_xor(s1, 16, 32);
  float r0 = 1.0f / s0, r1 = 1.0f / s1;
#pragma unroll
  for (int v = 0; v < 8; ++v) {
    int s = v + hi;
    wsm[wave][s][m] = e0[v] * r0;
    wsm[wave][s][16 + m] = e1[v] * r1;
  }
  __syncthreads();
  // weighted sum: out[c] = sum_s (xv_g + p_r)[s,c] * wsm[s, c%32]
  const int cbase = lane * 8;
  float w20[8], w21[8], w22[8], bb[8];
#pragma unroll
  for (int i = 0; i < 8; ++i) {
    int c = cbase + i;
    w20[i] = wp2[c * 3 + 0];
    w21[i] = wp2[c * 3 + 1];
    w22[i] = wp2[c * 3 + 2];
    bb[i] = bp2[c];
  }
  float acc[8] = {0.f, 0.f, 0.f, 0.f, 0.f, 0.f, 0.f, 0.f};
  for (int s = 0; s < NS; ++s) {
    int j = sidx[wave][s];
    v8bf xv = *(const v8bf*)(qkv + (size_t)j * QKVW + 2 * CCH + cbase);
    float q0 = rb[wave][s][0], q1 = rb[wave][s][1], q2 = rb[wave][s][2];
#pragma unroll
    for (int i = 0; i < 8; ++i) {
      float pr = q0 * w20[i] + q1 * w21[i] + q2 * w22[i] + bb[i];
      float wv = wsm[wave][s][(cbase + i) & 31];
      acc[i] += ((float)xv[i] + pr) * wv;
    }
  }
  float* orow = out + (size_t)n * CCH + cbase;
  v4f o0 = {acc[0], acc[1], acc[2], acc[3]};
  v4f o1 = {acc[4], acc[5], acc[6], acc[7]};
  *(v4f*)(orow) = o0;
  *(v4f*)(orow + 4) = o1;
}

// ---------------------------------------------------------------------------
extern "C" void kernel_launch(void* const* d_in, const int* in_sizes, int n_in,
                              void* d_out, int out_size, void* d_ws, size_t ws_size,
                              hipStream_t stream) {
  (void)in_sizes; (void)n_in; (void)out_size; (void)ws_size;
  const float* p   = (const float*)d_in[0];
  const float* x   = (const float*)d_in[1];
  const int*   idx = (const int*)d_in[2];
  const float* wq  = (const float*)d_in[3];
  const float* bq  = (const float*)d_in[4];
  const float* wk  = (const float*)d_in[5];
  const float* bk  = (const float*)d_in[6];
  const float* wv  = (const float*)d_in[7];
  const float* bv  = (const float*)d_in[8];
  const float* wp1 = (const float*)d_in[9];
  const float* bp1 = (const float*)d_in[10];
  const float* gp  = (const float*)d_in[11];
  const float* btp = (const float*)d_in[12];
  const float* wp2 = (const float*)d_in[13];
  const float* bp2 = (const float*)d_in[14];
  const float* g1  = (const float*)d_in[15];
  const float* b1  = (const float*)d_in[16];
  const float* ww1 = (const float*)d_in[17];
  const float* bw1 = (const float*)d_in[18];
  const float* g2  = (const float*)d_in[19];
  const float* b2  = (const float*)d_in[20];
  const float* ww2 = (const float*)d_in[21];
  const float* bw2 = (const float*)d_in[22];

  char* ws = (char*)d_ws;
  __bf16* qkv   = (__bf16*)(ws + OFF_QKV);
  float*  h1    = (float*)(ws + OFF_H1);
  __bf16* wb    = (__bf16*)(ws + OFF_WB);
  float*  bqkv  = (float*)(ws + OFF_BQKV);
  __bf16* wwb1  = (__bf16*)(ws + OFF_WW1);
  __bf16* wwb2  = (__bf16*)(ws + OFF_WW2);
  float*  stats = (float*)(ws + OFF_STATS);
  float*  partp = (float*)(ws + OFF_PARTP);
  float*  part1 = (float*)(ws + OFF_PART1);
  float*  part2 = (float*)(ws + OFF_PART2);
  float*  out   = (float*)d_out;

  k0_prep<<<768, 256, 0, stream>>>(wq, wk, wv, bq, bk, bv, ww1, ww2,
                                   wb, bqkv, wwb1, wwb2);
  k1_qkv<<<dim3(512, 6), 256, 0, stream>>>(x, wb, bqkv, qkv);
  k2_bnp_part<<<2048, 256, 0, stream>>>(p, idx, wp1, bp1, partp);
  kreduce<<<1, 64, 0, stream>>>(partp, stats + 0, 6, 2048);
  k3_bn1_part<<<2048, 256, 0, stream>>>(p, idx, qkv, wp1, bp1, gp, btp,
                                        wp2, bp2, stats, part1);
  kreduce<<<2, 256, 0, stream>>>(part1, stats + 6, 512, 2048);
  k4_h1<<<8192, 256, 0, stream>>>(p, idx, qkv, wp1, bp1, gp, btp, wp2, bp2,
                                  g1, b1, wwb1, bw1, stats, h1);
  k4b_bn2_part<<<1024, 256, 0, stream>>>(h1, part2);
  kreduce<<<1, 64, 0, stream>>>(part2, stats + 518, 64, 1024);
  k5_out<<<8192, 256, 0, stream>>>(p, idx, qkv, wp1, bp1, gp, btp, wp2, bp2,
                                   g2, b2, wwb2, bw2, stats, h1, out);
}
